// GIN_Critic_34187939676288
// MI455X (gfx1250) — compile-verified, matrix operand order checked by reading, not run on hardware
//
#include <hip/hip_runtime.h>
#include <hip/hip_bf16.h>

typedef _Float16 f16;
typedef __attribute__((ext_vector_type(16))) _Float16 v16h;
typedef __attribute__((ext_vector_type(8)))  _Float16 v8h;
typedef __attribute__((ext_vector_type(2)))  _Float16 v2h;
typedef __attribute__((ext_vector_type(8)))  float    v8f;

#define BB   8
#define NN   20000
#define EE   320000
#define BE   (BB * EE)        /* 2,560,000 total edges */
#define TWO_E (2 * EE)
#define NF   3
#define EMB  64
#define HID  128
#define TOTN (BB * NN)        /* 160,000 total nodes */

// ---------------------------------------------------------------------------
// WMMA fragment load: two contiguous 16-byte LDS reads -> 8 VGPRs, no repack.
// CDNA5 16-bit A layout (ISA 7.12.2): lane m=lane&15, kb=(lane>>4)*8 holds
// K in [kb, kb+8) (elems 0-7) and [kb+16, kb+24) (elems 8-15) of the 32-K tile.
// B uses the same striping with N in place of M, so we store weights as W^T.
// ---------------------------------------------------------------------------
union Frag { v16h v; v8h h[2]; };

__device__ __forceinline__ v16h frag_ld(const f16* p) {  // p must be 16B aligned
  Frag u;
  u.h[0] = *(const v8h*)(p);
  u.h[1] = *(const v8h*)(p + 16);
  return u.v;
}

// A tile: row-major f16 [16 x K] slab in LDS starting at `base`, pitch `ld`
__device__ __forceinline__ v16h frag_A(const f16* base, int ld, int k0, int lane) {
  return frag_ld(base + (lane & 15) * ld + k0 + ((lane >> 4) << 3));
}
// B tile: W^T layout [Nout x K] f16 in LDS, pitch ldk = K
__device__ __forceinline__ v16h frag_B(const f16* wT, int ldk, int k0, int n0, int lane) {
  return frag_ld(wT + (n0 + (lane & 15)) * ldk + k0 + ((lane >> 4) << 3));
}

__device__ __forceinline__ v8f wmma_f16(v16h a, v16h b, v8f c) {
  return __builtin_amdgcn_wmma_f32_16x16x32_f16(false, a, false, b, (short)0, c,
                                                false, false);
}

// ---------------------------------------------------------------------------
// Weight f32 -> f16 + transpose (stored as W^T so B fragments are contiguous)
//   w16[0]      : c1_w2^T  [EMB][HID]
//   w16[8192]   : c2_w1^T  [HID][EMB]
//   w16[16384]  : c2_w2^T  [EMB][HID]
// ---------------------------------------------------------------------------
__global__ void cvt_w_kernel(const float* __restrict__ c1w2,
                             const float* __restrict__ c2w1,
                             const float* __restrict__ c2w2,
                             f16* __restrict__ w16) {
  const int i = blockIdx.x * 256 + threadIdx.x;
  if (i >= HID * EMB) return;
  { const int r = i >> 6, c = i & 63;                       // [HID][EMB] src
    w16[c * HID + r] = (f16)c1w2[i]; }
  { const int r = i >> 7, c = i & 127;                      // [EMB][HID] src
    w16[HID * EMB + c * EMB + r] = (f16)c2w1[i]; }
  { const int r = i >> 6, c = i & 63;                       // [HID][EMB] src
    w16[2 * HID * EMB + c * HID + r] = (f16)c2w2[i]; }
}

// ---------------------------------------------------------------------------
// Build x[160k,3] = {action0, action1, node_feature}
// ---------------------------------------------------------------------------
__global__ void build_x_kernel(const float* __restrict__ actions,
                               const float* __restrict__ nodef,
                               float* __restrict__ x3) {
  const int i = blockIdx.x * 256 + threadIdx.x;   // i = b*N + n
  if (i < TOTN) {
    const float2 a = *(const float2*)(actions + i * 2);
    x3[i * 3 + 0] = a.x;
    x3[i * 3 + 1] = a.y;
    x3[i * 3 + 2] = nodef[i];
  }
}

// ---------------------------------------------------------------------------
// Conv1 scatter: agg1[dst] += x3[src]   (3 features, thread per edge)
// Replicates the reference's offset-then-reshape(2, B*E) indexing:
//   src[j] = ei_flat[j]    + (j/(2E))*N
//   dst[j] = ei_flat[BE+j] + ((BE+j)/(2E))*N
// ---------------------------------------------------------------------------
__global__ __launch_bounds__(256) void scatter1_kernel(
    const long long* __restrict__ ei, const float* __restrict__ x3,
    float* __restrict__ agg1) {
  const int j = blockIdx.x * 256 + threadIdx.x;
  if (j >= BE) return;
  const int s = (int)ei[j]      + (j / TWO_E) * NN;
  const int d = (int)ei[BE + j] + ((BE + j) / TWO_E) * NN;
  const float v0 = x3[s * 3 + 0];
  const float v1 = x3[s * 3 + 1];
  const float v2 = x3[s * 3 + 2];
  atomicAdd(&agg1[d * 3 + 0], v0);
  atomicAdd(&agg1[d * 3 + 1], v1);
  atomicAdd(&agg1[d * 3 + 2], v2);
}

// ---------------------------------------------------------------------------
// Conv1 MLP: x1 = relu( relu((x3+agg1) @ W1[3,128] + b1) @ W2[128,64] + b2 )
// 128 nodes / block (256 thr = 8 waves). Layer1 scalar (K=3), layer2 WMMA.
// ---------------------------------------------------------------------------
__global__ __launch_bounds__(256) void conv1_mlp_kernel(
    const float* __restrict__ x3, const float* __restrict__ agg1,
    const float* __restrict__ w1, const float* __restrict__ b1,
    const f16* __restrict__ w2T, const float* __restrict__ b2,
    float* __restrict__ x1) {
  __shared__ __attribute__((aligned(16))) float sh_h[128][4];
  __shared__ __attribute__((aligned(16))) float sw1[NF * HID];
  __shared__ __attribute__((aligned(16))) float sb1[HID];
  __shared__ __attribute__((aligned(16))) f16   sA[128][HID + 8];  // activations
  __shared__ __attribute__((aligned(16))) f16   sw2[EMB * HID];    // W2^T

  const int t     = threadIdx.x;
  const int node0 = blockIdx.x * 128;

  for (int i = t; i < NF * HID; i += 256) sw1[i] = w1[i];
  if (t < HID) sb1[t] = b1[t];
  { const unsigned* src = (const unsigned*)w2T;
    unsigned* dst = (unsigned*)sw2;
    for (int i = t; i < HID * EMB / 2; i += 256) dst[i] = src[i]; }
  for (int i = t; i < 128 * NF; i += 256) {
    const int node = i / NF, f = i % NF;
    const int g = (node0 + node) * NF + f;
    sh_h[node][f] = x3[g] + agg1[g];
  }
  __syncthreads();

  // layer 1: 128 nodes x 128 hidden, K = 3; two adjacent cols per thread
  for (int i = t; i < 128 * HID / 2; i += 256) {
    const int node = i >> 6;
    const int c    = (i & 63) << 1;
    float a0 = sb1[c], a1 = sb1[c + 1];
    const float h0 = sh_h[node][0], h1 = sh_h[node][1], h2 = sh_h[node][2];
    a0 += h0 * sw1[0 * HID + c]     + h1 * sw1[1 * HID + c]     + h2 * sw1[2 * HID + c];
    a1 += h0 * sw1[0 * HID + c + 1] + h1 * sw1[1 * HID + c + 1] + h2 * sw1[2 * HID + c + 1];
    v2h pk;
    pk[0] = (f16)fmaxf(a0, 0.0f);
    pk[1] = (f16)fmaxf(a1, 0.0f);
    *(v2h*)&sA[node][c] = pk;
  }
  __syncthreads();

  // layer 2: [128x128] @ [128x64] via WMMA; rowTile = wave, 4 col tiles each
  const int wave = t >> 5, lane = t & 31;
#pragma unroll
  for (int ct = 0; ct < 4; ++ct) {
    v8f c = {};
#pragma unroll
    for (int kk = 0; kk < HID; kk += 32) {
      const v16h a = frag_A(&sA[wave * 16][0], HID + 8, kk, lane);
      const v16h b = frag_B(sw2, HID, kk, ct * 16, lane);
      c = wmma_f16(a, b, c);
    }
    const int   col  = ct * 16 + (lane & 15);
    const int   r0   = wave * 16 + ((lane >> 4) << 3);
    const float bias = b2[col];
#pragma unroll
    for (int i = 0; i < 8; ++i) {
      const int node = node0 + r0 + i;
      x1[node * EMB + col] = fmaxf(c[i] + bias, 0.0f);
    }
  }
}

// ---------------------------------------------------------------------------
// Conv2 scatter: agg2[dst] += x1[src]  (64 feats, one wave32 per edge)
// ---------------------------------------------------------------------------
__global__ __launch_bounds__(256) void scatter2_kernel(
    const long long* __restrict__ ei, const float* __restrict__ x1,
    float* __restrict__ agg2) {
  const int wid  = (blockIdx.x * 256 + threadIdx.x) >> 5;
  const int lane = threadIdx.x & 31;
  if (wid >= BE) return;
  const int s = (int)ei[wid]      + (wid / TWO_E) * NN;
  const int d = (int)ei[BE + wid] + ((BE + wid) / TWO_E) * NN;
  const float v0 = x1[s * EMB + lane];
  const float v1 = x1[s * EMB + 32 + lane];
  atomicAdd(&agg2[d * EMB + lane],      v0);
  atomicAdd(&agg2[d * EMB + 32 + lane], v1);
}

// ---------------------------------------------------------------------------
// Conv2 MLP + fused sum-pool:
//   x2 = relu( relu((x1+agg2) @ W1[64,128] + b1) @ W2[128,64] + b2 )
//   pool[b,:] += column sums of x2 (x2 never touches HBM).
// 128 nodes / block; 16 | 20000 so every 16-row tile is batch-pure.
// ---------------------------------------------------------------------------
__global__ __launch_bounds__(256) void conv2_mlp_pool_kernel(
    const float* __restrict__ x1, const float* __restrict__ agg2,
    const f16* __restrict__ w1T, const float* __restrict__ b1,
    const f16* __restrict__ w2T, const float* __restrict__ b2,
    float* __restrict__ pool) {
  __shared__ __attribute__((aligned(16))) f16 sA[128][EMB + 8];   // layer-1 in
  __shared__ __attribute__((aligned(16))) f16 sH[128][HID + 8];   // layer-1 out
  __shared__ __attribute__((aligned(16))) f16 sw1[HID * EMB];     // W1^T
  __shared__ __attribute__((aligned(16))) f16 sw2[EMB * HID];     // W2^T

  const int t     = threadIdx.x;
  const int node0 = blockIdx.x * 128;

  { const unsigned* s1 = (const unsigned*)w1T;
    const unsigned* s2 = (const unsigned*)w2T;
    unsigned* d1 = (unsigned*)sw1;
    unsigned* d2 = (unsigned*)sw2;
    for (int i = t; i < HID * EMB / 2; i += 256) { d1[i] = s1[i]; d2[i] = s2[i]; } }
  for (int i = t; i < 128 * EMB / 2; i += 256) {   // two feats per thread
    const int node = i >> 5;
    const int c    = (i & 31) << 1;
    const int g = (node0 + node) * EMB + c;
    const float2 xv = *(const float2*)(x1 + g);
    const float2 av = *(const float2*)(agg2 + g);
    v2h pk;
    pk[0] = (f16)(xv.x + av.x);
    pk[1] = (f16)(xv.y + av.y);
    *(v2h*)&sA[node][c] = pk;
  }
  __syncthreads();

  const int wave = t >> 5, lane = t & 31;

  // layer 1: [128x64] @ [64x128], rowTile = wave, 8 col tiles each (K = 64)
#pragma unroll
  for (int ct = 0; ct < 8; ++ct) {
    v8f c = {};
#pragma unroll
    for (int kk = 0; kk < EMB; kk += 32) {
      const v16h a = frag_A(&sA[wave * 16][0], EMB + 8, kk, lane);
      const v16h b = frag_B(sw1, EMB, kk, ct * 16, lane);
      c = wmma_f16(a, b, c);
    }
    const int   col  = ct * 16 + (lane & 15);
    const int   r0   = wave * 16 + ((lane >> 4) << 3);
    const float bias = b1[col];
#pragma unroll
    for (int i = 0; i < 8; ++i)
      sH[r0 + i][col] = (f16)fmaxf(c[i] + bias, 0.0f);
  }
  __syncthreads();

  // layer 2: [128x128] @ [128x64] (K = 128) + in-register pooling
  const int batch = (node0 + wave * 16) / NN;   // tile is batch-pure
#pragma unroll
  for (int ct = 0; ct < 4; ++ct) {
    v8f c = {};
#pragma unroll
    for (int kk = 0; kk < HID; kk += 32) {
      const v16h a = frag_A(&sH[wave * 16][0], HID + 8, kk, lane);
      const v16h b = frag_B(sw2, HID, kk, ct * 16, lane);
      c = wmma_f16(a, b, c);
    }
    const int   col  = ct * 16 + (lane & 15);
    const float bias = b2[col];
    float s = 0.0f;
#pragma unroll
    for (int i = 0; i < 8; ++i) s += fmaxf(c[i] + bias, 0.0f);
    s += __shfl_xor(s, 16, 32);                  // rows 0-7 + rows 8-15
    if (lane < 16) atomicAdd(&pool[batch * EMB + col], s);
  }
}

// ---------------------------------------------------------------------------
// Head MLP: hmid = relu(pool[8,64] @ mlp_w[64,128] + mlp_b)
// ---------------------------------------------------------------------------
__global__ void head_kernel(const float* __restrict__ pool,
                            const float* __restrict__ mlp_w,
                            const float* __restrict__ mlp_b,
                            float* __restrict__ hmid) {
  const int t = threadIdx.x;   // 128 threads
#pragma unroll
  for (int b = 0; b < BB; ++b) {
    float acc = mlp_b[t];
    for (int k = 0; k < EMB; ++k) acc += pool[b * EMB + k] * mlp_w[k * HID + t];
    hmid[b * HID + t] = fmaxf(acc, 0.0f);
  }
}

// ---------------------------------------------------------------------------
// Output: out[b,n] = sigmoid(hmid[b,:] @ out_w[:,n] + out_b[n])
// Thread per column; out_w streamed exactly once (10.24 MB).
// ---------------------------------------------------------------------------
__global__ __launch_bounds__(256) void out_kernel(
    const float* __restrict__ hmid, const float* __restrict__ out_w,
    const float* __restrict__ out_b, float* __restrict__ out) {
  __shared__ float sh[BB * HID];
  const int t = threadIdx.x;
  for (int i = t; i < BB * HID; i += 256) sh[i] = hmid[i];
  __syncthreads();
  const int n = blockIdx.x * 256 + t;
  if (n >= NN) return;
  float acc[BB];
  const float bias = out_b[n];
#pragma unroll
  for (int b = 0; b < BB; ++b) acc[b] = bias;
  for (int k = 0; k < HID; ++k) {
    const float w = out_w[k * NN + n];
#pragma unroll
    for (int b = 0; b < BB; ++b) acc[b] += sh[b * HID + k] * w;
  }
#pragma unroll
  for (int b = 0; b < BB; ++b)
    out[b * NN + n] = 1.0f / (1.0f + __expf(-acc[b]));
}

// ---------------------------------------------------------------------------
extern "C" void kernel_launch(void* const* d_in, const int* in_sizes, int n_in,
                              void* d_out, int out_size, void* d_ws, size_t ws_size,
                              hipStream_t stream) {
  const float*     actions = (const float*)d_in[0];
  const float*     nodef   = (const float*)d_in[1];
  const long long* ei      = (const long long*)d_in[2];   // int64 per reference
  const float* c1_w1 = (const float*)d_in[3];
  const float* c1_b1 = (const float*)d_in[4];
  const float* c1_w2 = (const float*)d_in[5];
  const float* c1_b2 = (const float*)d_in[6];
  const float* c2_w1 = (const float*)d_in[7];
  const float* c2_b1 = (const float*)d_in[8];
  const float* c2_w2 = (const float*)d_in[9];
  const float* c2_b2 = (const float*)d_in[10];
  const float* mlp_w = (const float*)d_in[11];
  const float* mlp_b = (const float*)d_in[12];
  const float* out_w = (const float*)d_in[13];
  const float* out_b = (const float*)d_in[14];
  float* out = (float*)d_out;

  // workspace layout (all offsets 256B-aligned)
  char* ws = (char*)d_ws;
  size_t off = 0;
  float* x3   = (float*)(ws + off); off += (size_t)TOTN * 3 * 4;          // 1.92 MB
  float* agg1 = (float*)(ws + off); off += (size_t)TOTN * 3 * 4;          // 1.92 MB
  float* x1   = (float*)(ws + off); off += (size_t)TOTN * EMB * 4;        // 40.96 MB
  float* agg2 = (float*)(ws + off); off += (size_t)TOTN * EMB * 4;        // 40.96 MB
  float* pool = (float*)(ws + off); off += 2048;
  float* hmid = (float*)(ws + off); off += 4096;
  f16*   w16  = (f16*)(ws + off);   off += 3 * (size_t)HID * EMB * 2;
  f16* w2_c1T = w16;                     // [EMB][HID]
  f16* w1_c2T = w16 + HID * EMB;         // [HID][EMB]
  f16* w2_c2T = w16 + 2 * HID * EMB;     // [EMB][HID]

  hipMemsetAsync(agg1, 0, (size_t)TOTN * 3 * 4, stream);
  hipMemsetAsync(agg2, 0, (size_t)TOTN * EMB * 4, stream);
  hipMemsetAsync(pool, 0, BB * EMB * 4, stream);

  cvt_w_kernel<<<(HID * EMB + 255) / 256, 256, 0, stream>>>(c1_w2, c2_w1, c2_w2, w16);
  build_x_kernel<<<(TOTN + 255) / 256, 256, 0, stream>>>(actions, nodef, x3);
  scatter1_kernel<<<(BE + 255) / 256, 256, 0, stream>>>(ei, x3, agg1);
  conv1_mlp_kernel<<<TOTN / 128, 256, 0, stream>>>(x3, agg1, c1_w1, c1_b1,
                                                   w2_c1T, c1_b2, x1);
  scatter2_kernel<<<BE / 8, 256, 0, stream>>>(ei, x1, agg2);
  conv2_mlp_pool_kernel<<<TOTN / 128, 256, 0, stream>>>(x1, agg2, w1_c2T, c2_b1,
                                                        w2_c2T, c2_b2, pool);
  head_kernel<<<1, 128, 0, stream>>>(pool, mlp_w, mlp_b, hmid);
  out_kernel<<<(NN + 255) / 256, 256, 0, stream>>>(hmid, out_w, out_b, out);
}
